// MaxwellGNNBlock_86088324481807
// MI455X (gfx1250) — matrix operand order repaired and checked by reading, not compiled, not run on hardware
//
#include <hip/hip_runtime.h>
#include <hip/hip_bf16.h>
#include <math.h>

// ---------------------------------------------------------------------------
// MaxwellGNNBlock for MI455X (gfx1250, wave32).
//
// - Edge GEMM folded into node GEMM (hWe = h_mod @ We[0:128]) => all dense
//   math is N x 128 x {128,256} GEMMs on the WMMA matrix cores
//   (v_wmma_f32_16x16x32_f16, f16 in / f32 accumulate).
// - GEMM kernel: one wave owns a 16x64 output strip (4 WMMA accumulators),
//   reusing the A fragment across 4 column tiles -> 4 back-to-back WMMAs per
//   K-step for better XDL occupancy and 4x less A traffic.
// - Edge phase (bandwidth-bound): wave-per-edge float4 gathers, shfl head
//   reductions, 3-pass segment softmax with f32 atomics.
// ---------------------------------------------------------------------------

typedef __attribute__((ext_vector_type(16))) _Float16 v16h;
typedef __attribute__((ext_vector_type(8)))  _Float16 v8h;
typedef __attribute__((ext_vector_type(8)))  float    v8f;

#define D128 128
#define NHEAD 4

static __device__ __forceinline__ float4 ld4(const float* p) {
    return *(const float4*)p;
}

static __device__ __forceinline__ void atomicMaxF(float* addr, float val) {
    unsigned int* ua = (unsigned int*)addr;
    unsigned int old = *ua;
    while (true) {
        float f = __uint_as_float(old);
        if (f >= val) break;
        unsigned int assumed = old;
        old = atomicCAS(ua, assumed, __float_as_uint(val));
        if (old == assumed) break;
    }
}

// ------------------------- tiny style GEMM (1x128 @ 128x256) ----------------
__global__ void k_style(const float* __restrict__ t_emb,
                        const float* __restrict__ fc_W,
                        const float* __restrict__ fc_b,
                        float* __restrict__ style) {
    int j = threadIdx.x;          // 0..255
    float acc = fc_b[j];
    #pragma unroll 8
    for (int k = 0; k < D128; ++k)
        acc += t_emb[k] * fc_W[k * 256 + j];
    style[j] = acc;
}

// ---------------- weight transpose + fp32 -> f16 conversion ------------------
// dst is (M x K) row-major f16, src is (K x M) row-major f32.
__global__ void k_wtrans(const float* __restrict__ src,
                         _Float16* __restrict__ dst, int K, int M) {
    int idx = blockIdx.x * blockDim.x + threadIdx.x;
    if (idx >= K * M) return;
    int m = idx / K;
    int k = idx - m * K;
    dst[idx] = (_Float16)src[k * M + m];
}

// --------------------------- AdaGN (wave per node) --------------------------
__global__ void k_adagn(const float* __restrict__ h_source,
                        const float* __restrict__ gn_w,
                        const float* __restrict__ gn_b,
                        const float* __restrict__ style,
                        _Float16* __restrict__ hmod16, int N) {
    int node = blockIdx.x * 8 + (threadIdx.x >> 5);
    int lane = threadIdx.x & 31;
    if (node >= N) return;
    float4 x = ld4(h_source + (size_t)node * D128 + lane * 4);
    float s  = x.x + x.y + x.z + x.w;
    float ss = x.x * x.x + x.y * x.y + x.z * x.z + x.w * x.w;
    // group of 16 features = 4 consecutive lanes
    s  += __shfl_xor(s, 1, 32);  ss += __shfl_xor(ss, 1, 32);
    s  += __shfl_xor(s, 2, 32);  ss += __shfl_xor(ss, 2, 32);
    float mu  = s * (1.0f / 16.0f);
    float var = ss * (1.0f / 16.0f) - mu * mu;
    float inv = rsqrtf(var + 1e-5f);
    float v[4] = {x.x, x.y, x.z, x.w};
    #pragma unroll
    for (int j = 0; j < 4; ++j) {
        int f = lane * 4 + j;
        float nv = (v[j] - mu) * inv * gn_w[f] + gn_b[f];
        float out = nv * (1.0f + style[f]) + style[D128 + f];
        hmod16[(size_t)node * D128 + f] = (_Float16)out;
    }
}

// -------------- WMMA GEMM: C = A @ Bt^T + bias (+resid), 16x64/wave ---------
// A: (Nrows x K) f16 row-major.  Bt: (M x K) f16 row-major (pre-transposed W).
// One wave computes a 16x64 strip: A fragment reused across 4 column tiles,
// 4 independent WMMAs per K-step (K stepped by 32).
__global__ void k_gemm(const _Float16* __restrict__ A,
                       const _Float16* __restrict__ Bt,
                       const float* __restrict__ bias,
                       const float* __restrict__ resid,
                       float* __restrict__ C,
                       int Nrows, int K, int M) {
    int wid  = blockIdx.x * (blockDim.x >> 5) + (threadIdx.x >> 5);
    int lane = threadIdx.x & 31;
    int strips = M >> 6;                 // 64-column strips
    int rowT = (Nrows + 15) >> 4;
    if (wid >= rowT * strips) return;
    int row_t = wid / strips;
    int strip = wid - row_t * strips;

    const _Float16* Arow = A  + (size_t)(row_t * 16 + (lane & 15)) * K;
    const _Float16* B0   = Bt + (size_t)(strip * 64 + (lane & 15)) * K;
    int khA = (lane >> 4) * 8;    // A: hi half-wave carries K+8..15 / K+24..31
    int khB = (lane >> 4) * 8;    // B: hi half-wave carries K+8..15 / K+24..31

    v8f acc0 = {}, acc1 = {}, acc2 = {}, acc3 = {};
    for (int k0 = 0; k0 < K; k0 += 32) {
        v8h a0 = *(const v8h*)(Arow + k0 + khA);
        v8h a1 = *(const v8h*)(Arow + k0 + 16 + khA);
        v16h a;
        #pragma unroll
        for (int i = 0; i < 8; ++i) { a[i] = a0[i]; a[8 + i] = a1[i]; }

        v16h b;
        v8h b0, b1;
        #define LOADB(t)                                                   \
            b0 = *(const v8h*)(B0 + (size_t)(t) * 16 * K + k0 + khB);      \
            b1 = *(const v8h*)(B0 + (size_t)(t) * 16 * K + k0 + khB + 16); \
            _Pragma("unroll")                                              \
            for (int i = 0; i < 8; ++i) { b[i] = b0[i]; b[8 + i] = b1[i]; }

        LOADB(0)
        acc0 = __builtin_amdgcn_wmma_f32_16x16x32_f16(false, a, false, b,
                                                      (short)0, acc0, false, false);
        LOADB(1)
        acc1 = __builtin_amdgcn_wmma_f32_16x16x32_f16(false, a, false, b,
                                                      (short)0, acc1, false, false);
        LOADB(2)
        acc2 = __builtin_amdgcn_wmma_f32_16x16x32_f16(false, a, false, b,
                                                      (short)0, acc2, false, false);
        LOADB(3)
        acc3 = __builtin_amdgcn_wmma_f32_16x16x32_f16(false, a, false, b,
                                                      (short)0, acc3, false, false);
        #undef LOADB
    }

    int rbase = row_t * 16 + (lane >> 4) * 8;
    v8f accs[4] = {acc0, acc1, acc2, acc3};
    #pragma unroll
    for (int t = 0; t < 4; ++t) {
        int col = strip * 64 + t * 16 + (lane & 15);
        float bv = bias ? bias[col] : 0.0f;
        #pragma unroll
        for (int r = 0; r < 8; ++r) {
            size_t idx = (size_t)(rbase + r) * M + col;
            float v = accs[t][r] + bv;
            if (resid) v += resid[idx];
            C[idx] = v;
        }
    }
}

// --------------------------- softmax state init -----------------------------
__global__ void k_init_softmax(float* __restrict__ amax,
                               float* __restrict__ denom, int n) {
    int i = blockIdx.x * blockDim.x + threadIdx.x;
    if (i >= n) return;
    amax[i]  = -3.402823466e38f;
    denom[i] = 0.0f;
}

__global__ void k_zero(float* __restrict__ p, int n) {
    int i = blockIdx.x * blockDim.x + threadIdx.x;
    if (i < n) p[i] = 0.0f;
}

// ----------------- edge pass 1: logits + running segment max ----------------
__global__ void k_edge1(const long long* __restrict__ ei,
                        const float* __restrict__ pos,
                        const float* __restrict__ xl,
                        const float* __restrict__ xr,
                        const float* __restrict__ hwe,
                        const float* __restrict__ We,     // fp32 (130x128)
                        const float* __restrict__ att,    // (4x32) flat
                        float* __restrict__ logit,        // (E x 4)
                        float* __restrict__ amax, long long E) {
    long long e = (long long)blockIdx.x * 8 + (threadIdx.x >> 5);
    int lane = threadIdx.x & 31;
    if (e >= E) return;
    long long src = ei[e];
    long long dst = ei[E + e];

    float rx = pos[src * 2]     - pos[dst * 2];
    float ry = pos[src * 2 + 1] - pos[dst * 2 + 1];
    float inv_d2 = 1.0f / (rx * rx + ry * ry + 1e-8f);
    float gx = rx * inv_d2, gy = ry * inv_d2;
    float o0 = -gy, o1 = gx;                 // ortho = (-geom.y, geom.x)

    int f = lane * 4;
    float4 xls = ld4(xl  + (size_t)src * D128 + f);
    float4 xrd = ld4(xr  + (size_t)dst * D128 + f);
    float4 ws  = ld4(hwe + (size_t)src * D128 + f);
    float4 wd  = ld4(hwe + (size_t)dst * D128 + f);
    float4 wo0 = ld4(We + 128 * D128 + f);
    float4 wo1 = ld4(We + 129 * D128 + f);
    float4 at  = ld4(att + f);

    float xv[4];
    xv[0] = xls.x + xrd.x + (ws.x - wd.x) + o0 * wo0.x + o1 * wo1.x;
    xv[1] = xls.y + xrd.y + (ws.y - wd.y) + o0 * wo0.y + o1 * wo1.y;
    xv[2] = xls.z + xrd.z + (ws.z - wd.z) + o0 * wo0.z + o1 * wo1.z;
    xv[3] = xls.w + xrd.w + (ws.w - wd.w) + o0 * wo0.w + o1 * wo1.w;
    float av[4] = {at.x, at.y, at.z, at.w};
    float p = 0.0f;
    #pragma unroll
    for (int j = 0; j < 4; ++j) {
        float x = xv[j];
        x = (x > 0.0f) ? x : 0.2f * x;       // leaky_relu(0.2)
        p += x * av[j];
    }
    // reduce over 8 lanes (one head = 32 channels = 8 lanes)
    p += __shfl_xor(p, 1, 32);
    p += __shfl_xor(p, 2, 32);
    p += __shfl_xor(p, 4, 32);
    if ((lane & 7) == 0) {
        int h = lane >> 3;
        logit[e * NHEAD + h] = p;
        atomicMaxF(&amax[dst * NHEAD + h], p);
    }
}

// -------------- edge pass 2: exp(logit - max), accumulate denom -------------
__global__ void k_edge2(const long long* __restrict__ ei,
                        float* __restrict__ logit,       // in: logit, out: ex
                        const float* __restrict__ amax,
                        float* __restrict__ denom, long long E) {
    long long idx = (long long)blockIdx.x * blockDim.x + threadIdx.x;
    if (idx >= E * NHEAD) return;
    long long e = idx >> 2;
    int h = (int)(idx & 3);
    long long dst = ei[E + e];
    float ex = expf(logit[idx] - amax[dst * NHEAD + h]);
    logit[idx] = ex;
    atomicAdd(&denom[dst * NHEAD + h], ex);
}

// -------------- edge pass 3: msg = x_l[src] * alpha, segment sum ------------
__global__ void k_edge3(const long long* __restrict__ ei,
                        const float* __restrict__ xl,
                        const float* __restrict__ ex,     // (E x 4)
                        const float* __restrict__ denom,
                        float* __restrict__ aggr, long long E) {
    long long e = (long long)blockIdx.x * 8 + (threadIdx.x >> 5);
    int lane = threadIdx.x & 31;
    if (e >= E) return;
    long long src = ei[e];
    long long dst = ei[E + e];
    int h = lane >> 3;
    float alpha = ex[e * NHEAD + h] / (denom[dst * NHEAD + h] + 1e-16f);
    int f = lane * 4;
    float4 m = ld4(xl + (size_t)src * D128 + f);
    float* out = aggr + (size_t)dst * D128 + f;
    atomicAdd(out + 0, m.x * alpha);
    atomicAdd(out + 1, m.y * alpha);
    atomicAdd(out + 2, m.z * alpha);
    atomicAdd(out + 3, m.w * alpha);
}

// ------------------- aggr + gat_b  ->  f16 (GEMM A operand) -----------------
__global__ void k_biascvt(const float* __restrict__ aggr,
                          const float* __restrict__ gat_b,
                          _Float16* __restrict__ out16, int n) {
    int i = blockIdx.x * blockDim.x + threadIdx.x;
    if (i >= n) return;
    out16[i] = (_Float16)(aggr[i] + gat_b[i & (D128 - 1)]);
}

// ----------------------- LayerNorm (wave per node) -> f16 -------------------
__global__ void k_ln(const float* __restrict__ x,
                     const float* __restrict__ ln_w,
                     const float* __restrict__ ln_b,
                     _Float16* __restrict__ out16, int N) {
    int node = blockIdx.x * 8 + (threadIdx.x >> 5);
    int lane = threadIdx.x & 31;
    if (node >= N) return;
    float4 v = ld4(x + (size_t)node * D128 + lane * 4);
    float s  = v.x + v.y + v.z + v.w;
    float ss = v.x * v.x + v.y * v.y + v.z * v.z + v.w * v.w;
    #pragma unroll
    for (int w = 1; w < 32; w <<= 1) {
        s  += __shfl_xor(s, w, 32);
        ss += __shfl_xor(ss, w, 32);
    }
    float mu  = s * (1.0f / 128.0f);
    float var = ss * (1.0f / 128.0f) - mu * mu;
    float inv = rsqrtf(var + 1e-5f);
    float vv[4] = {v.x, v.y, v.z, v.w};
    #pragma unroll
    for (int j = 0; j < 4; ++j) {
        int f = lane * 4 + j;
        out16[(size_t)node * D128 + f] =
            (_Float16)((vv[j] - mu) * inv * ln_w[f] + ln_b[f]);
    }
}

// ----------------------------- exact GELU -> f16 ----------------------------
__global__ void k_gelu(const float* __restrict__ z,
                       _Float16* __restrict__ out16, int n) {
    int i = blockIdx.x * blockDim.x + threadIdx.x;
    if (i >= n) return;
    float x = z[i];
    out16[i] = (_Float16)(0.5f * x * (1.0f + erff(x * 0.70710678118654752f)));
}

// ============================================================================
extern "C" void kernel_launch(void* const* d_in, const int* in_sizes, int n_in,
                              void* d_out, int out_size, void* d_ws, size_t ws_size,
                              hipStream_t stream) {
    const float*     h_target = (const float*)d_in[0];
    const float*     h_source = (const float*)d_in[1];
    const long long* ei       = (const long long*)d_in[2];
    const float*     pos      = (const float*)d_in[3];
    const float*     t_emb    = (const float*)d_in[4];
    const float*     gn_w     = (const float*)d_in[5];
    const float*     gn_b     = (const float*)d_in[6];
    const float*     fc_W     = (const float*)d_in[7];
    const float*     fc_b     = (const float*)d_in[8];
    const float*     Wl       = (const float*)d_in[9];
    const float*     bl       = (const float*)d_in[10];
    const float*     Wr       = (const float*)d_in[11];
    const float*     br       = (const float*)d_in[12];
    const float*     We       = (const float*)d_in[13];   // (130,128)
    const float*     att      = (const float*)d_in[14];   // (4,32)
    const float*     gat_b    = (const float*)d_in[15];
    const float*     Wp       = (const float*)d_in[16];
    const float*     bp       = (const float*)d_in[17];
    const float*     ln_w     = (const float*)d_in[18];
    const float*     ln_b     = (const float*)d_in[19];
    const float*     W1       = (const float*)d_in[20];   // (128,256)
    const float*     b1       = (const float*)d_in[21];
    const float*     W2       = (const float*)d_in[22];   // (256,128)
    const float*     b2       = (const float*)d_in[23];
    float*           out      = (float*)d_out;

    const int       N = in_sizes[0] / D128;
    const long long E = (long long)in_sizes[2] / 2;

    // ---- workspace layout (256B aligned) ----
    char*  base = (char*)d_ws;
    size_t off  = 0;
    auto alloc = [&](size_t bytes) -> char* {
        char* p = base + off;
        off = (off + bytes + 255) & ~(size_t)255;
        return p;
    };
    float*    style   = (float*)   alloc(256 * sizeof(float));
    _Float16* wl_t    = (_Float16*)alloc((size_t)128 * 128 * 2);
    _Float16* wr_t    = (_Float16*)alloc((size_t)128 * 128 * 2);
    _Float16* we_t    = (_Float16*)alloc((size_t)128 * 128 * 2);
    _Float16* wp_t    = (_Float16*)alloc((size_t)128 * 128 * 2);
    _Float16* w1_t    = (_Float16*)alloc((size_t)256 * 128 * 2);
    _Float16* w2_t    = (_Float16*)alloc((size_t)128 * 256 * 2);
    _Float16* hmod16  = (_Float16*)alloc((size_t)N * D128 * 2);
    float*    xl      = (float*)   alloc((size_t)N * D128 * 4);
    float*    xr_aggr = (float*)   alloc((size_t)N * D128 * 4);  // x_r, then aggr
    float*    hwe_tmp = (float*)   alloc((size_t)N * D128 * 4);  // hWe, then post-Wp
    float*    logit   = (float*)   alloc((size_t)E * NHEAD * 4); // logit, then ex
    float*    amax    = (float*)   alloc((size_t)N * NHEAD * 4);
    float*    denom   = (float*)   alloc((size_t)N * NHEAD * 4);
    _Float16* f16buf  = (_Float16*)alloc((size_t)N * 256 * 2);   // aggr16/ln16/g16
    float*    zbuf    = (float*)   alloc((size_t)N * 256 * 4);
    (void)ws_size; (void)n_in; (void)out_size;

    const int T = 256;
    auto grid1 = [](long long n, int t) { return (unsigned)((n + t - 1) / t); };
    // GEMM grids: waves = rowTiles * (M/64) strips, 8 waves/block
    auto ggrid = [](int rows, int M) {
        long long w = (long long)((rows + 15) / 16) * (M / 64);
        return (unsigned)((w + 7) / 8);
    };
    unsigned edgeBlocks = (unsigned)((E + 7) / 8);   // wave per edge

    // 1) style = t_emb @ fc_W + fc_b
    k_style<<<1, 256, 0, stream>>>(t_emb, fc_W, fc_b, style);

    // 2) transpose + f16-convert all GEMM weights (Bt layout: M x K)
    k_wtrans<<<grid1(128 * 128, T), T, 0, stream>>>(Wl, wl_t, 128, 128);
    k_wtrans<<<grid1(128 * 128, T), T, 0, stream>>>(Wr, wr_t, 128, 128);
    k_wtrans<<<grid1(128 * 128, T), T, 0, stream>>>(We, we_t, 128, 128); // rows 0..127
    k_wtrans<<<grid1(128 * 128, T), T, 0, stream>>>(Wp, wp_t, 128, 128);
    k_wtrans<<<grid1(128 * 256, T), T, 0, stream>>>(W1, w1_t, 128, 256); // -> (256x128)
    k_wtrans<<<grid1(256 * 128, T), T, 0, stream>>>(W2, w2_t, 256, 128); // -> (128x256)

    // 3) AdaGN -> hmod16 (f16 GEMM A operand)
    k_adagn<<<grid1((long long)N * 32, T), T, 0, stream>>>(
        h_source, gn_w, gn_b, style, hmod16, N);

    // 4) node GEMMs on WMMA: x_l, x_r, hWe
    k_gemm<<<ggrid(N, 128), T, 0, stream>>>(hmod16, wl_t, bl, nullptr, xl,      N, 128, 128);
    k_gemm<<<ggrid(N, 128), T, 0, stream>>>(hmod16, wr_t, br, nullptr, xr_aggr, N, 128, 128);
    k_gemm<<<ggrid(N, 128), T, 0, stream>>>(hmod16, we_t, nullptr, nullptr, hwe_tmp, N, 128, 128);

    // 5) segment softmax over edges (3 passes)
    k_init_softmax<<<grid1((long long)N * NHEAD, T), T, 0, stream>>>(amax, denom, N * NHEAD);
    k_edge1<<<edgeBlocks, T, 0, stream>>>(ei, pos, xl, xr_aggr, hwe_tmp, We, att,
                                          logit, amax, E);
    k_zero<<<grid1((long long)N * D128, T), T, 0, stream>>>(xr_aggr, N * D128); // -> aggr
    k_edge2<<<grid1(E * NHEAD, T), T, 0, stream>>>(ei, logit, amax, denom, E);
    k_edge3<<<edgeBlocks, T, 0, stream>>>(ei, xl, logit, denom, xr_aggr, E);

    // 6) post-GAT: (+gat_b) @ Wp + bp -> LN -> @W1 + b1 -> GELU -> @W2 + b2 + h_target
    k_biascvt<<<grid1((long long)N * D128, T), T, 0, stream>>>(xr_aggr, gat_b, f16buf, N * D128);
    k_gemm<<<ggrid(N, 128), T, 0, stream>>>(f16buf, wp_t, bp, nullptr, hwe_tmp, N, 128, 128);
    k_ln<<<grid1((long long)N * 32, T), T, 0, stream>>>(hwe_tmp, ln_w, ln_b, f16buf, N);
    k_gemm<<<ggrid(N, 256), T, 0, stream>>>(f16buf, w1_t, b1, nullptr, zbuf, N, 128, 256);
    k_gelu<<<grid1((long long)N * 256, T), T, 0, stream>>>(zbuf, f16buf, N * 256);
    k_gemm<<<ggrid(N, 128), T, 0, stream>>>(f16buf, w2_t, b2, h_target, out, N, 256, 128);
}
